// QLlamaDecoderLayer_68238440399485
// MI455X (gfx1250) — matrix operand order repaired and checked by reading, not compiled, not run on hardware
//
#include <hip/hip_runtime.h>

#define SQ    1024
#define BB    2
#define NHEAD 32
#define NKH   8
#define DH    128
#define HIDD  4096
#define INTERD 11008
#define NTOK  (BB*SQ)      /* 2048 */
#define EPSF  1e-5f
#define THETAF 10000.f

typedef __attribute__((ext_vector_type(16))) __bf16 v16bf;
typedef __attribute__((ext_vector_type(8)))  __bf16 v8bf;
typedef __attribute__((ext_vector_type(8)))  float  v8f;
typedef __attribute__((ext_vector_type(4)))  int    v4i;
typedef __attribute__((address_space(1))) v4i gv4i;   // global int4
typedef __attribute__((address_space(3))) v4i lv4i;   // LDS int4

#if defined(__has_builtin)
#if __has_builtin(__builtin_amdgcn_global_load_async_to_lds_b128) && \
    __has_builtin(__builtin_amdgcn_s_wait_asynccnt)
#define USE_ASYNC_LDS 1
#endif
#endif

// 16-byte global -> LDS copy (async DMA on CDNA5 if available)
static __device__ inline void cp16_g2l(const __bf16* g, __bf16* l) {
#ifdef USE_ASYNC_LDS
  __builtin_amdgcn_global_load_async_to_lds_b128((gv4i*)g, (lv4i*)l, 0, 0);
#else
  uint4 v = *(const uint4*)g;
  *(uint4*)l = v;
#endif
}
template <int N>
static __device__ inline void async_wait() {
#ifdef USE_ASYNC_LDS
  __builtin_amdgcn_s_wait_asynccnt(N);
#endif
}

static __device__ inline v8f wmma_bf16(v16bf a, v16bf b, v8f c) {
  return __builtin_amdgcn_wmma_f32_16x16x32_bf16(false, a, false, b, (short)0, c, false, false);
}

// A-fragment (16x32 bf16): lane holds row (lane&15); K runs {kp..kp+7, kp+16..kp+23}, kp=(lane>>4)*8
static __device__ inline v16bf ld_afrag(const __bf16* tile, int lane, int stride) {
  const __bf16* p = tile + (lane & 15) * stride + ((lane >> 4) << 3);
  v8bf lo = *(const v8bf*)p;
  v8bf hi = *(const v8bf*)(p + 16);
  v16bf a;
#pragma unroll
  for (int i = 0; i < 8; ++i) { a[i] = lo[i]; a[i + 8] = hi[i]; }
  return a;
}
// B-fragment (32x16 bf16, [k][n] tile): lane holds column N=(lane&15); K runs {kp..kp+7, kp+16..kp+23}, kp=(lane>>4)*8
static __device__ inline v16bf ld_bfrag(const __bf16* tile, int lane, int stride) {
  const __bf16* p = tile + ((lane >> 4) << 3) * stride + (lane & 15);
  v16bf b;
#pragma unroll
  for (int i = 0; i < 8; ++i) { b[i] = p[i * stride]; b[i + 8] = p[(i + 16) * stride]; }
  return b;
}

// --------------------------------------------- weight f32->bf16 + transpose
// W[R][C] f32  ->  Wt[C][R] bf16   (R,C multiples of 32)
__global__ __launch_bounds__(256)
void wt_conv_kern(const float* __restrict__ W, __bf16* __restrict__ Wt,
                  int R, int C) {
  __shared__ float tile[32][33];
  int rb = blockIdx.x * 32, cb = blockIdx.y * 32;
  int t = threadIdx.x;
  int r = t >> 3, c4 = (t & 7) * 4;
  float4 f = *(const float4*)(W + (size_t)(rb + r) * C + cb + c4);
  tile[r][c4 + 0] = f.x; tile[r][c4 + 1] = f.y;
  tile[r][c4 + 2] = f.z; tile[r][c4 + 3] = f.w;
  __syncthreads();
  int cc = t >> 3, r4 = (t & 7) * 4;
  union { __bf16 h[4]; uint2 u; } pk;
  pk.h[0] = (__bf16)tile[r4 + 0][cc];
  pk.h[1] = (__bf16)tile[r4 + 1][cc];
  pk.h[2] = (__bf16)tile[r4 + 2][cc];
  pk.h[3] = (__bf16)tile[r4 + 3][cc];
  *(uint2*)(Wt + (size_t)(cb + cc) * R + rb + r4) = pk.u;
}

// ---------------------------------------------------------------- rmsnorm
__global__ __launch_bounds__(256)
void rmsnorm_kern(const float* __restrict__ X, const float* __restrict__ W,
                  float* __restrict__ Y) {
  __shared__ float red[256];
  int row = blockIdx.x, tid = threadIdx.x;
  const float* x = X + (size_t)row * HIDD;
  float ss = 0.f;
  for (int i = tid; i < HIDD; i += 256) { float v = x[i]; ss += v * v; }
  red[tid] = ss; __syncthreads();
  for (int s2 = 128; s2 > 0; s2 >>= 1) {
    if (tid < s2) red[tid] += red[tid + s2];
    __syncthreads();
  }
  float inv = rsqrtf(red[0] / (float)HIDD + EPSF);
  float* y = Y + (size_t)row * HIDD;
  for (int i = tid; i < HIDD; i += 256) y[i] = x[i] * inv * W[i];
}

// ------------------------------------------------- reorder grouped fake-quant
__global__ __launch_bounds__(128)
void act_quant_kern(const float* __restrict__ X, const int* __restrict__ ri,
                    __bf16* __restrict__ Y, int C) {
  __shared__ float smn[128], smx[128];
  int tid = threadIdx.x, g = blockIdx.x, row = blockIdx.y;
  int c = ri[g * 128 + tid];
  float v = X[(size_t)row * C + c];
  smn[tid] = v; smx[tid] = v; __syncthreads();
  for (int s2 = 64; s2 > 0; s2 >>= 1) {
    if (tid < s2) {
      smn[tid] = fminf(smn[tid], smn[tid + s2]);
      smx[tid] = fmaxf(smx[tid], smx[tid + s2]);
    }
    __syncthreads();
  }
  float wmin = smn[0], wmax = smx[0];
  int bits = (g == 0) ? 8 : (g < 9 ? 6 : 4);
  float qmax = (float)((1 << bits) - 1);
  float sc = fmaxf(wmax - wmin, 1e-5f) / qmax;
  float base = fminf(fmaxf(rintf(-wmin / sc), 0.f), qmax);
  float q = (fminf(fmaxf(rintf(v / sc) + base, 0.f), qmax) - base) * sc;
  Y[(size_t)row * C + c] = (__bf16)q;
}

// ------------------------------------------------------------- WMMA GEMM
// C[M,N] = A_bf16[M,K] @ Wt_bf16[K][N] (+ residual). BK=64, double-buffered
// async LDS staging. M,N mult of 128; K mult of 64.
__global__ __launch_bounds__(256)
void gemm_bf16(const __bf16* __restrict__ A, const __bf16* __restrict__ Wt,
               float* __restrict__ Cout, const float* __restrict__ res,
               int M, int N, int K) {
  __shared__ __align__(16) __bf16 As[2][128 * 72];   // [m][k] pad 64->72
  __shared__ __align__(16) __bf16 Bs[2][64 * 136];   // [k][n] pad 128->136
  int t = threadIdx.x, lane = t & 31, wave = t >> 5;
  int wm = wave >> 2, wn = wave & 3;
  int rowBase = blockIdx.y * 128, colBase = blockIdx.x * 128;
  v8f z = {0.f, 0.f, 0.f, 0.f, 0.f, 0.f, 0.f, 0.f};
  v8f acc[4][2];
#pragma unroll
  for (int i = 0; i < 4; ++i)
#pragma unroll
    for (int j = 0; j < 2; ++j) acc[i][j] = z;

  int ar = t >> 1, ak = (t & 1) << 5;
  int br = t >> 2, bn = (t & 3) << 5;
  const __bf16* gA = A + (size_t)(rowBase + ar) * K + ak;
  const __bf16* gB = Wt + (size_t)br * N + colBase + bn;

  auto stage = [&](int bi, int k0) {
    const __bf16* ga = gA + k0;
    const __bf16* gb = gB + (size_t)k0 * N;
    __bf16* la = &As[bi][ar * 72 + ak];
    __bf16* lb = &Bs[bi][br * 136 + bn];
#pragma unroll
    for (int j = 0; j < 4; ++j) cp16_g2l(ga + j * 8, la + j * 8);
#pragma unroll
    for (int j = 0; j < 4; ++j) cp16_g2l(gb + j * 8, lb + j * 8);
    __builtin_prefetch(ga + 64, 0, 1);                 // tile+2 hint
    __builtin_prefetch(gb + (size_t)64 * N, 0, 1);
  };
  auto compute = [&](int bi) {
#pragma unroll
    for (int kk = 0; kk < 64; kk += 32) {
      v16bf af[4], bfr[2];
#pragma unroll
      for (int mt = 0; mt < 4; ++mt)
        af[mt] = ld_afrag(&As[bi][(wm * 64 + mt * 16) * 72 + kk], lane, 72);
#pragma unroll
      for (int nt = 0; nt < 2; ++nt)
        bfr[nt] = ld_bfrag(&Bs[bi][kk * 136 + wn * 32 + nt * 16], lane, 136);
#pragma unroll
      for (int mt = 0; mt < 4; ++mt)
#pragma unroll
        for (int nt = 0; nt < 2; ++nt)
          acc[mt][nt] = wmma_bf16(af[mt], bfr[nt], acc[mt][nt]);
    }
  };

  int nsteps = K >> 6;
  stage(0, 0);
  for (int s = 0; s < nsteps; ++s) {
    int bi = s & 1;
    if (s + 1 < nsteps) {
      stage(1 - bi, (s + 1) << 6);   // overlap next tile DMA with compute
      async_wait<8>();               // current tile's 8 copies complete
    } else {
      async_wait<0>();
    }
    __syncthreads();
    compute(bi);
    __syncthreads();
  }

  int mh = (lane >> 4) << 3, nn = lane & 15;
#pragma unroll
  for (int mt = 0; mt < 4; ++mt)
#pragma unroll
    for (int nt = 0; nt < 2; ++nt)
#pragma unroll
      for (int r = 0; r < 8; ++r) {
        int m = rowBase + wm * 64 + mt * 16 + r + mh;
        int n = colBase + wn * 32 + nt * 16 + nn;
        float v = acc[mt][nt][r];
        if (res) v += res[(size_t)m * N + n];
        Cout[(size_t)m * N + n] = v;
      }
}

// ------------------------------------------------------------- RoPE on Q
__global__ __launch_bounds__(64)
void rope_q_kern(float* __restrict__ Q, const int* __restrict__ pids) {
  int h = blockIdx.x, tok = blockIdx.y, i = threadIdx.x;
  float pos = (float)pids[tok % SQ];
  float* p = Q + (size_t)tok * HIDD + h * DH;
  float invf = __expf(-((float)(2 * i) / (float)DH) * __logf(THETAF));
  float ang = pos * invf;
  float cs = __cosf(ang), sn = __sinf(ang);
  float x1 = p[i], x2 = p[i + 64];
  p[i]      = x1 * cs - x2 * sn;
  p[i + 64] = x2 * cs + x1 * sn;
}

// --------------------------------------- KV: 4-bit per-head fake-quant, K-RoPE
__global__ __launch_bounds__(128)
void kvprep_kern(float* __restrict__ K, float* __restrict__ V,
                 const int* __restrict__ pids) {
  __shared__ float smn[128], smx[128], sval[128];
  int kh = blockIdx.x, tok = blockIdx.y, tid = threadIdx.x;
  float* kp = K + (size_t)tok * (NKH * DH) + kh * DH;
  float* vp = V + (size_t)tok * (NKH * DH) + kh * DH;
  float kv = kp[tid];
  smn[tid] = kv; smx[tid] = kv; __syncthreads();
  for (int s2 = 64; s2 > 0; s2 >>= 1) {
    if (tid < s2) { smn[tid] = fminf(smn[tid], smn[tid + s2]); smx[tid] = fmaxf(smx[tid], smx[tid + s2]); }
    __syncthreads();
  }
  {
    float sc = fmaxf(smx[0] - smn[0], 1e-5f) / 15.f;
    float base = fminf(fmaxf(rintf(-smn[0] / sc), 0.f), 15.f);
    sval[tid] = (fminf(fmaxf(rintf(kv / sc) + base, 0.f), 15.f) - base) * sc;
  }
  __syncthreads();
  if (tid < 64) {
    float pos = (float)pids[tok % SQ];
    float invf = __expf(-((float)(2 * tid) / (float)DH) * __logf(THETAF));
    float ang = pos * invf, cs = __cosf(ang), sn = __sinf(ang);
    float x1 = sval[tid], x2 = sval[tid + 64];
    kp[tid]      = x1 * cs - x2 * sn;
    kp[tid + 64] = x2 * cs + x1 * sn;
  }
  __syncthreads();
  float vv = vp[tid];
  smn[tid] = vv; smx[tid] = vv; __syncthreads();
  for (int s2 = 64; s2 > 0; s2 >>= 1) {
    if (tid < s2) { smn[tid] = fminf(smn[tid], smn[tid + s2]); smx[tid] = fmaxf(smx[tid], smx[tid + s2]); }
    __syncthreads();
  }
  float sc2 = fmaxf(smx[0] - smn[0], 1e-5f) / 15.f;
  float base2 = fminf(fmaxf(rintf(-smn[0] / sc2), 0.f), 15.f);
  vp[tid] = (fminf(fmaxf(rintf(vv / sc2) + base2, 0.f), 15.f) - base2) * sc2;
}

// -------------------------------- pack K (transposed) and V to bf16 for attn
// K f32 [tok][kh*DH+d] -> Ktr bf16 [b*NKH+kh][d][s]
// V f32 [tok][kh*DH+d] -> Vbf bf16 [b*NKH+kh][s][d]
__global__ __launch_bounds__(256)
void kv_pack_kern(const float* __restrict__ K, const float* __restrict__ V,
                  __bf16* __restrict__ Ktr, __bf16* __restrict__ Vbf) {
  __shared__ float tile[32][132];
  int bh2 = blockIdx.x;
  int b = bh2 >> 3, kh = bh2 & 7;
  int sb = blockIdx.y * 32;
  int t = threadIdx.x;
  int si = t >> 3, d0 = (t & 7) << 4;
  const float* kp = K + (size_t)(b * SQ + sb + si) * (NKH * DH) + kh * DH + d0;
#pragma unroll
  for (int i = 0; i < 16; i += 4) {
    float4 f = *(const float4*)(kp + i);
    tile[si][d0 + i + 0] = f.x; tile[si][d0 + i + 1] = f.y;
    tile[si][d0 + i + 2] = f.z; tile[si][d0 + i + 3] = f.w;
  }
  __syncthreads();
  int d = t >> 1, s0 = (t & 1) << 4;
  __bf16* outp = Ktr + ((size_t)bh2 * DH + d) * SQ + sb + s0;
#pragma unroll
  for (int w = 0; w < 2; ++w) {
    union { __bf16 h[8]; uint4 u; } pk;
#pragma unroll
    for (int i = 0; i < 8; ++i) pk.h[i] = (__bf16)tile[s0 + w * 8 + i][d];
    *(uint4*)(outp + w * 8) = pk.u;
  }
  const float* vp = V + (size_t)(b * SQ + sb + si) * (NKH * DH) + kh * DH + d0;
  __bf16* vout = Vbf + ((size_t)bh2 * SQ + sb + si) * DH + d0;
#pragma unroll
  for (int w = 0; w < 2; ++w) {
    union { __bf16 h[8]; uint4 u; } pv;
#pragma unroll
    for (int i = 0; i < 8; i += 4) {
      float4 f = *(const float4*)(vp + w * 8 + i);
      pv.h[i + 0] = (__bf16)f.x; pv.h[i + 1] = (__bf16)f.y;
      pv.h[i + 2] = (__bf16)f.z; pv.h[i + 3] = (__bf16)f.w;
    }
    *(uint4*)(vout + w * 8) = pv.u;
  }
}

// ----------------------------------------------- flash attention (WMMA bf16)
// Q f32 [tok][h*DH+d]; Ktr bf16 [b*NKH+kh][d][s]; Vbf bf16 [b*NKH+kh][s][d]
__global__ __launch_bounds__(256)
void attention_kern(const float* __restrict__ Q, const __bf16* __restrict__ Ktr,
                    const __bf16* __restrict__ Vbf, float* __restrict__ O) {
  __shared__ __align__(16) __bf16 Kt[2][128 * 40];   // [d][kv] padded (32->40)
  __shared__ __align__(16) __bf16 Vs[2][32 * 128];   // [kv][d]
  __shared__ __align__(16) __bf16 Ps[8 * 16 * 32];   // per-wave P tile
  int t = threadIdx.x, lane = t & 31, wave = t >> 5;
  int blk = blockIdx.x;
  int qblk = blk & 7, bh = blk >> 3;
  int b = bh >> 5, h = bh & 31, kh = h >> 2;
  int bh2 = b * NKH + kh;
  int qbase = qblk * 128, qr0 = qbase + wave * 16;
  const float scale = 0.08838834764831845f;          // 1/sqrt(128)
  int mh = (lane >> 4) << 3, nn = lane & 15;

  // Q fragments (16 rows x 128 d -> 4 A-frags)
  int srow0 = qr0 + (lane & 15);
  const float* qp = Q + (size_t)(b * SQ + srow0) * HIDD + h * DH + ((lane >> 4) << 3);
  v16bf qf[4];
#pragma unroll
  for (int c = 0; c < 4; ++c) {
    float4 a0 = *(const float4*)(qp + c * 32);
    float4 a1 = *(const float4*)(qp + c * 32 + 4);
    float4 b0 = *(const float4*)(qp + c * 32 + 16);
    float4 b1 = *(const float4*)(qp + c * 32 + 20);
    qf[c][0]  = (__bf16)a0.x; qf[c][1]  = (__bf16)a0.y; qf[c][2]  = (__bf16)a0.z; qf[c][3]  = (__bf16)a0.w;
    qf[c][4]  = (__bf16)a1.x; qf[c][5]  = (__bf16)a1.y; qf[c][6]  = (__bf16)a1.z; qf[c][7]  = (__bf16)a1.w;
    qf[c][8]  = (__bf16)b0.x; qf[c][9]  = (__bf16)b0.y; qf[c][10] = (__bf16)b0.z; qf[c][11] = (__bf16)b0.w;
    qf[c][12] = (__bf16)b1.x; qf[c][13] = (__bf16)b1.y; qf[c][14] = (__bf16)b1.z; qf[c][15] = (__bf16)b1.w;
  }

  v8f z = {0.f, 0.f, 0.f, 0.f, 0.f, 0.f, 0.f, 0.f};
  v8f ot[8];
#pragma unroll
  for (int u = 0; u < 8; ++u) ot[u] = z;
  float rm[8], rs[8];
#pragma unroll
  for (int r = 0; r < 8; ++r) { rm[r] = -3.0e38f; rs[r] = 0.f; }

  const __bf16* kbase = Ktr + (size_t)bh2 * DH * SQ;
  const __bf16* vbase = Vbf + (size_t)bh2 * SQ * DH;
  // staging: Kt: thread covers d = t>>1, kv-span (t&1)*16..+15 (2 chunks)
  //          Vs: thread covers kv = t>>3, d-span  (t&7)*16..+15 (2 chunks)
  auto stageA = [&](int bi, int kvb) {
    const __bf16* gk = kbase + (size_t)(t >> 1) * SQ + kvb + ((t & 1) << 4);
    __bf16* lk = &Kt[bi][(t >> 1) * 40 + ((t & 1) << 4)];
    cp16_g2l(gk, lk);
    cp16_g2l(gk + 8, lk + 8);
    const __bf16* gv = vbase + (size_t)(kvb + (t >> 3)) * DH + ((t & 7) << 4);
    __bf16* lv = &Vs[bi][(t >> 3) * 128 + ((t & 7) << 4)];
    cp16_g2l(gv, lv);
    cp16_g2l(gv + 8, lv + 8);
  };

  int jmax = qbase / 32 + 4;
  stageA(0, 0);
  for (int j = 0; j < jmax; ++j) {
    int kvb = j * 32, bi = j & 1;
    if (j + 1 < jmax) {
      stageA(1 - bi, (j + 1) * 32);   // overlap next chunk DMA
      async_wait<4>();
    } else {
      async_wait<0>();
    }
    __syncthreads();
    // S = Q @ K^T : 2 kv-col tiles, 4 k-chunks over d
    v8f sc0 = z, sc1 = z;
#pragma unroll
    for (int c = 0; c < 4; ++c) {
      v16bf b0 = ld_bfrag(&Kt[bi][(c * 32) * 40 + 0], lane, 40);
      v16bf b1 = ld_bfrag(&Kt[bi][(c * 32) * 40 + 16], lane, 40);
      sc0 = wmma_bf16(qf[c], b0, sc0);
      sc1 = wmma_bf16(qf[c], b1, sc1);
    }
    // online softmax + store P (bf16) into wave-private LDS
    __bf16* Pw = &Ps[wave * 512];
#pragma unroll
    for (int r = 0; r < 8; ++r) {
      int m = r + mh, srow = qr0 + m;
      float s0 = (kvb + nn      <= srow) ? sc0[r] * scale : -3.0e38f;
      float s1 = (kvb + 16 + nn <= srow) ? sc1[r] * scale : -3.0e38f;
      float mx = fmaxf(s0, s1);
#pragma unroll
      for (int o = 1; o < 16; o <<= 1) mx = fmaxf(mx, __shfl_xor(mx, o, 32));
      float newm = fmaxf(rm[r], mx);
      float f  = (rm[r] < -1.0e37f) ? 0.f : __expf(rm[r] - newm);
      float p0 = (s0    < -1.0e37f) ? 0.f : __expf(s0 - newm);
      float p1 = (s1    < -1.0e37f) ? 0.f : __expf(s1 - newm);
      float ps = p0 + p1;
#pragma unroll
      for (int o = 1; o < 16; o <<= 1) ps += __shfl_xor(ps, o, 32);
      rs[r] = rs[r] * f + ps;
      rm[r] = newm;
#pragma unroll
      for (int u = 0; u < 8; ++u) ot[u][r] *= f;
      Pw[m * 32 + nn]      = (__bf16)p0;
      Pw[m * 32 + 16 + nn] = (__bf16)p1;
    }
    // O += P(16x32) @ V(32x128) : 8 d-tiles
    v16bf pf = ld_afrag(Pw, lane, 32);
#pragma unroll
    for (int u = 0; u < 8; ++u) {
      v16bf vf = ld_bfrag(&Vs[bi][u * 16], lane, 128);
      ot[u] = wmma_bf16(pf, vf, ot[u]);
    }
    __syncthreads();
  }
#pragma unroll
  for (int u = 0; u < 8; ++u)
#pragma unroll
    for (int r = 0; r < 8; ++r) {
      int m = r + mh;
      int tok = b * SQ + qr0 + m;
      O[(size_t)tok * HIDD + h * DH + u * 16 + nn] = ot[u][r] / rs[r];
    }
}

// ------------------------------------------------------------- SiLU * up
__global__ void silu_mul_kern(float* __restrict__ G, const float* __restrict__ U, long n) {
  long i = (long)blockIdx.x * blockDim.x + threadIdx.x;
  if (i < n) { float g = G[i]; G[i] = (g / (1.f + __expf(-g))) * U[i]; }
}

// --------------------------------------------------------------- launcher
extern "C" void kernel_launch(void* const* d_in, const int* in_sizes, int n_in,
                              void* d_out, int out_size, void* d_ws, size_t ws_size,
                              hipStream_t stream) {
  const float* hidden = (const float*)d_in[0];
  const int*   pids   = (const int*)d_in[1];
  const float* wq     = (const float*)d_in[2];
  const float* wk     = (const float*)d_in[3];
  const float* wv     = (const float*)d_in[4];
  const float* wo     = (const float*)d_in[5];
  const float* wgate  = (const float*)d_in[6];
  const float* wup    = (const float*)d_in[7];
  const float* wdown  = (const float*)d_in[8];
  const float* ln1    = (const float*)d_in[9];
  const float* ln2    = (const float*)d_in[10];
  const int* ri_qkv   = (const int*)d_in[11];
  const int* ri_o     = (const int*)d_in[12];
  const int* ri_mlp   = (const int*)d_in[13];
  const int* ri_down  = (const int*)d_in[14];

  char* p = (char*)d_ws;
  auto alloc = [&](size_t bytes) { char* r = p; p += (bytes + 255) & ~(size_t)255; return r; };
  const size_t SZ_H  = (size_t)NTOK * HIDD;
  const size_t SZ_KV = (size_t)NTOK * (NKH * DH);
  const size_t SZ_I  = (size_t)NTOK * INTERD;

  float*  xnorm = (float*)alloc(SZ_H * 4);
  __bf16* aq    = (__bf16*)alloc(SZ_I * 2);
  float*  q     = (float*)alloc(SZ_H * 4);
  float*  k     = (float*)alloc(SZ_KV * 4);
  float*  v     = (float*)alloc(SZ_KV * 4);
  float*  attn  = (float*)alloc(SZ_H * 4);
  float*  hbuf  = (float*)alloc(SZ_H * 4);
  float*  g     = (float*)alloc(SZ_I * 4);
  float*  u     = (float*)alloc(SZ_I * 4);
  __bf16* ktr   = (__bf16*)alloc(SZ_KV * 2);
  __bf16* vbf   = (__bf16*)alloc(SZ_KV * 2);
  __bf16* wtq   = (__bf16*)alloc((size_t)HIDD * HIDD * 2);
  __bf16* wtk   = (__bf16*)alloc((size_t)HIDD * (NKH * DH) * 2);
  __bf16* wtv   = (__bf16*)alloc((size_t)HIDD * (NKH * DH) * 2);
  __bf16* wto   = (__bf16*)alloc((size_t)HIDD * HIDD * 2);
  __bf16* wtg   = (__bf16*)alloc((size_t)HIDD * INTERD * 2);
  __bf16* wtu   = (__bf16*)alloc((size_t)HIDD * INTERD * 2);
  __bf16* wtd   = (__bf16*)alloc((size_t)INTERD * HIDD * 2);
  float*  out   = (float*)d_out;

  // 0) convert+transpose weights to bf16 Wt[K][N]
  wt_conv_kern<<<dim3(HIDD/32,   HIDD/32),  256, 0, stream>>>(wq,    wtq, HIDD,    HIDD);
  wt_conv_kern<<<dim3((NKH*DH)/32, HIDD/32),256, 0, stream>>>(wk,    wtk, NKH*DH,  HIDD);
  wt_conv_kern<<<dim3((NKH*DH)/32, HIDD/32),256, 0, stream>>>(wv,    wtv, NKH*DH,  HIDD);
  wt_conv_kern<<<dim3(HIDD/32,   HIDD/32),  256, 0, stream>>>(wo,    wto, HIDD,    HIDD);
  wt_conv_kern<<<dim3(INTERD/32, HIDD/32),  256, 0, stream>>>(wgate, wtg, INTERD,  HIDD);
  wt_conv_kern<<<dim3(INTERD/32, HIDD/32),  256, 0, stream>>>(wup,   wtu, INTERD,  HIDD);
  wt_conv_kern<<<dim3(HIDD/32,   INTERD/32),256, 0, stream>>>(wdown, wtd, HIDD,    INTERD);

  // 1) rmsnorm + quant
  rmsnorm_kern<<<NTOK, 256, 0, stream>>>(hidden, ln1, xnorm);
  act_quant_kern<<<dim3(HIDD/128, NTOK), 128, 0, stream>>>(xnorm, ri_qkv, aq, HIDD);
  // 2) QKV
  gemm_bf16<<<dim3(HIDD/128, NTOK/128), 256, 0, stream>>>(aq, wtq, q, nullptr, NTOK, HIDD, HIDD);
  gemm_bf16<<<dim3((NKH*DH)/128, NTOK/128), 256, 0, stream>>>(aq, wtk, k, nullptr, NTOK, NKH*DH, HIDD);
  gemm_bf16<<<dim3((NKH*DH)/128, NTOK/128), 256, 0, stream>>>(aq, wtv, v, nullptr, NTOK, NKH*DH, HIDD);
  // 3) KV quant + RoPE, then pack to bf16 (K transposed)
  kvprep_kern<<<dim3(NKH, NTOK), 128, 0, stream>>>(k, v, pids);
  rope_q_kern<<<dim3(NHEAD, NTOK), 64, 0, stream>>>(q, pids);
  kv_pack_kern<<<dim3(BB * NKH, SQ / 32), 256, 0, stream>>>(k, v, ktr, vbf);
  // 4) attention
  attention_kern<<<BB * NHEAD * (SQ/128), 256, 0, stream>>>(q, ktr, vbf, attn);
  // 5) O projection + residual
  act_quant_kern<<<dim3(HIDD/128, NTOK), 128, 0, stream>>>(attn, ri_o, aq, HIDD);
  gemm_bf16<<<dim3(HIDD/128, NTOK/128), 256, 0, stream>>>(aq, wto, hbuf, hidden, NTOK, HIDD, HIDD);
  // 6) MLP
  rmsnorm_kern<<<NTOK, 256, 0, stream>>>(hbuf, ln2, xnorm);
  act_quant_kern<<<dim3(HIDD/128, NTOK), 128, 0, stream>>>(xnorm, ri_mlp, aq, HIDD);
  gemm_bf16<<<dim3(INTERD/128, NTOK/128), 256, 0, stream>>>(aq, wtg, g, nullptr, NTOK, INTERD, HIDD);
  gemm_bf16<<<dim3(INTERD/128, NTOK/128), 256, 0, stream>>>(aq, wtu, u, nullptr, NTOK, INTERD, HIDD);
  silu_mul_kern<<<(int)((SZ_I + 255) / 256), 256, 0, stream>>>(g, u, (long)SZ_I);
  act_quant_kern<<<dim3(INTERD/128, NTOK), 128, 0, stream>>>(g, ri_down, aq, INTERD);
  gemm_bf16<<<dim3(HIDD/128, NTOK/128), 256, 0, stream>>>(aq, wtd, out, hbuf, NTOK, HIDD, INTERD);
}
